// TransHyperScale_32744830664767
// MI455X (gfx1250) — matrix operand order repaired and checked by reading, not compiled, not run on hardware
//
#include <hip/hip_runtime.h>
#include <cstdint>
#include <cstddef>

// ---------------------------------------------------------------------------
// TransHyperScale forward for gfx1250 (MI455X): all contractions via
// v_wmma_f32_16x16x32_f16; GEMM tiles staged to LDS with
// global_load_async_to_lds_b128 (ASYNCcnt); flash-style attention with
// on-the-fly contextual relative-position bias gathered from qEh/qEw.
// ---------------------------------------------------------------------------

typedef __attribute__((ext_vector_type(16))) _Float16 v16h;
typedef __attribute__((ext_vector_type(8)))  _Float16 v8h;
typedef __attribute__((ext_vector_type(4)))  _Float16 v4h;
typedef __attribute__((ext_vector_type(8)))  float    v8f;
typedef __attribute__((ext_vector_type(4)))  float    v4f;

#define B_C     4
#define CIN_C   384
#define COUT_C  192
#define DIM_C   384
#define HEADS_C 6
#define DH_C    64
#define INNER_C 384
#define MLP_C   1536

static __device__ inline v8f vzero8() {
  v8f r;
#pragma unroll
  for (int i = 0; i < 8; ++i) r[i] = 0.0f;
  return r;
}

static __device__ inline float gelu_tanh(float x) {
  const float k0 = 0.7978845608028654f;   // sqrt(2/pi)
  const float k1 = 0.044715f;
  float t = tanhf(k0 * (x + k1 * x * x * x));
  return 0.5f * x * (1.0f + t);
}

// Issue one async 16-byte global->LDS copy for this lane.
static __device__ inline void async_copy_b128(const void* gsrc, void* ldst) {
  asm volatile("global_load_async_to_lds_b128 %0, %1, off"
               :: "v"((uint32_t)(uintptr_t)ldst),
                  "v"((uint64_t)(uintptr_t)gsrc)
               : "memory");
}
static __device__ inline void wait_async0() {
  asm volatile("s_wait_asynccnt 0x0" ::: "memory");
}

// --- fragment gathers -------------------------------------------------------
// A-fragment (16x32 f16, MxK) from f32 LDS tile [m][k], stride in floats.
// lane<16: M=lane, K = 0..15 ; lane>=16: same M, K = 16..31.
static __device__ inline v16h ldsA_f32(const float* base, int stride) {
  int lane = threadIdx.x & 31;
  int hf = lane >> 4, m = lane & 15;
  const float* p = base + m * stride + 16 * hf;
  v4f a = *(const v4f*)p;
  v4f b = *(const v4f*)(p + 4);
  v4f c = *(const v4f*)(p + 8);
  v4f d = *(const v4f*)(p + 12);
  v16h r;
#pragma unroll
  for (int i = 0; i < 4; ++i) {
    r[i]      = (_Float16)a[i];
    r[4 + i]  = (_Float16)b[i];
    r[8 + i]  = (_Float16)c[i];
    r[12 + i] = (_Float16)d[i];
  }
  return r;
}

// B-fragment (32x16 f16, KxN) from f32 LDS tile stored [n][k].
static __device__ inline v16h ldsB_f32(const float* base, int stride) {
  int lane = threadIdx.x & 31;
  int hf = lane >> 4, n = lane & 15;
  const float* p = base + n * stride + 16 * hf;
  v16h r;
#pragma unroll
  for (int i = 0; i < 4; ++i) {
    v4f f = *(const v4f*)(p + 4 * i);
#pragma unroll
    for (int j = 0; j < 4; ++j) r[4 * i + j] = (_Float16)f[j];
  }
  return r;
}

// A-fragment from f16 LDS tile [m][k], stride in halves.
static __device__ inline v16h ldsA_f16(const _Float16* base, int stride) {
  int lane = threadIdx.x & 31;
  int hf = lane >> 4, m = lane & 15;
  const _Float16* p = base + m * stride + 16 * hf;
  v8h lo = *(const v8h*)p;
  v8h hi = *(const v8h*)(p + 8);
  v16h r;
#pragma unroll
  for (int i = 0; i < 8; ++i) { r[i] = lo[i]; r[8 + i] = hi[i]; }
  return r;
}

// B-fragment from f16 LDS tile stored [n][k], stride in halves.
static __device__ inline v16h ldsB_f16(const _Float16* base, int stride) {
  int lane = threadIdx.x & 31;
  int hf = lane >> 4, n = lane & 15;
  const _Float16* p = base + n * stride + 16 * hf;
  v8h lo = *(const v8h*)p;
  v8h hi = *(const v8h*)(p + 8);
  v16h r;
#pragma unroll
  for (int i = 0; i < 8; ++i) { r[i] = lo[i]; r[8 + i] = hi[i]; }
  return r;
}

// ---------------------------------------------------------------------------
// Tiled WMMA GEMM:  C[M,N] = epilogue(alpha * A[M,K] @ B + bias) (+ residual)
// BT=false: B row-major [K,N]; BT=true: B row-major [N,K] (e.g. conv weights).
// Block = 128 threads (4 waves), block tile 64x64, K step 32.
// A tile (and BT B tile) staged via async global->LDS copies (f32 in LDS).
// ---------------------------------------------------------------------------
template <bool BT>
__global__ void __launch_bounds__(128)
gemm_wmma_kernel(const float* __restrict__ A, const float* __restrict__ B,
                 float* C, const float* __restrict__ bias, const float* res,
                 int M, int N, int K, float alpha, int act) {
  __shared__ float    Asf[64 * 36];                 // A tile, f32, stride 36
  __shared__ float    Bsf[BT ? 64 * 36 : 4];        // BT: B tile, f32
  __shared__ _Float16 BsH[BT ? 8 : 64 * 40];        // !BT: B tile, f16 [n][k]

  const int tid  = threadIdx.x;
  const int lane = tid & 31;
  const int wave = tid >> 5;
  const int hf   = lane >> 4;
  const int l16  = lane & 15;
  const int row0 = blockIdx.y * 64;
  const int col0 = blockIdx.x * 64;
  const int wr   = (wave >> 1) * 32;
  const int wc   = (wave & 1) * 32;

  v8f acc[2][2];
#pragma unroll
  for (int i = 0; i < 2; ++i)
#pragma unroll
    for (int j = 0; j < 2; ++j) acc[i][j] = vzero8();

  for (int k0 = 0; k0 < K; k0 += 32) {
    // --- async stage A tile (64 rows x 32 f32 = 512 x 16B chunks) ---
    {
      const int ch0 = tid * 4;
#pragma unroll
      for (int i = 0; i < 4; ++i) {
        int ch = ch0 + i;
        int r = ch >> 3, off = (ch & 7) * 4;        // off in floats
        async_copy_b128(A + (size_t)(row0 + r) * K + k0 + off,
                        &Asf[r * 36 + off]);
      }
    }
    if constexpr (BT) {
      // --- async stage B tile ([n][k] already in memory) ---
      const int ch0 = tid * 4;
#pragma unroll
      for (int i = 0; i < 4; ++i) {
        int ch = ch0 + i;
        int r = ch >> 3, off = (ch & 7) * 4;
        async_copy_b128(B + (size_t)(col0 + r) * K + k0 + off,
                        &Bsf[r * 36 + off]);
      }
    } else {
      // --- VGPR transpose-stage B tile into BsH[n][k] (f16) ---
      const int kk = tid & 31;
      const int np = (tid >> 5) * 16;
      const float* src = B + (size_t)(k0 + kk) * N + col0 + np;
#pragma unroll
      for (int i = 0; i < 4; ++i) {
        v4f f = *(const v4f*)(src + 4 * i);
#pragma unroll
        for (int j = 0; j < 4; ++j)
          BsH[(np + 4 * i + j) * 40 + kk] = (_Float16)f[j];
      }
      if (k0 + 32 < K)
        __builtin_prefetch(B + (size_t)(k0 + 32 + kk) * N + col0, 0, 1);
    }
    wait_async0();
    __syncthreads();

    v16h a0 = ldsA_f32(&Asf[(wr) * 36], 36);
    v16h a1 = ldsA_f32(&Asf[(wr + 16) * 36], 36);
    v16h b0, b1;
    if constexpr (BT) {
      b0 = ldsB_f32(&Bsf[(wc) * 36], 36);
      b1 = ldsB_f32(&Bsf[(wc + 16) * 36], 36);
    } else {
      b0 = ldsB_f16(&BsH[(wc) * 40], 40);
      b1 = ldsB_f16(&BsH[(wc + 16) * 40], 40);
    }

    acc[0][0] = __builtin_amdgcn_wmma_f32_16x16x32_f16(false, a0, false, b0,
                                                       (short)0, acc[0][0], false, false);
    acc[0][1] = __builtin_amdgcn_wmma_f32_16x16x32_f16(false, a0, false, b1,
                                                       (short)0, acc[0][1], false, false);
    acc[1][0] = __builtin_amdgcn_wmma_f32_16x16x32_f16(false, a1, false, b0,
                                                       (short)0, acc[1][0], false, false);
    acc[1][1] = __builtin_amdgcn_wmma_f32_16x16x32_f16(false, a1, false, b1,
                                                       (short)0, acc[1][1], false, false);
    __syncthreads();
  }

  // --- epilogue (C layout: element e -> row e + 8*hf, col l16) ---
#pragma unroll
  for (int i = 0; i < 2; ++i) {
#pragma unroll
    for (int j = 0; j < 2; ++j) {
      const int cbase = col0 + wc + 16 * j + l16;
      float bv = (bias != nullptr) ? bias[cbase] : 0.0f;
#pragma unroll
      for (int e = 0; e < 8; ++e) {
        int r = row0 + wr + 16 * i + e + 8 * hf;
        float v = alpha * acc[i][j][e] + bv;
        if (act == 1) v = gelu_tanh(v);
        if (res != nullptr) v += res[(size_t)r * N + cbase];
        C[(size_t)r * N + cbase] = v;
      }
    }
  }
}

// ---------------------------------------------------------------------------
// LayerNorm over last dim (=384). One wave per row.
// ---------------------------------------------------------------------------
__global__ void __launch_bounds__(128)
ln_fwd_kernel(const float* __restrict__ x, float* __restrict__ y,
              const float* __restrict__ w, const float* __restrict__ b, int M) {
  int row  = blockIdx.x * 4 + (threadIdx.x >> 5);
  int lane = threadIdx.x & 31;
  if (row >= M) return;
  const float* xr = x + (size_t)row * DIM_C;
  float v[12];
  float s = 0.0f;
#pragma unroll
  for (int i = 0; i < 12; ++i) { v[i] = xr[lane + 32 * i]; s += v[i]; }
#pragma unroll
  for (int o = 1; o < 32; o <<= 1) s += __shfl_xor(s, o, 32);
  float mean = s * (1.0f / 384.0f);
  float s2 = 0.0f;
#pragma unroll
  for (int i = 0; i < 12; ++i) { float d = v[i] - mean; s2 += d * d; }
#pragma unroll
  for (int o = 1; o < 32; o <<= 1) s2 += __shfl_xor(s2, o, 32);
  float inv = rsqrtf(s2 * (1.0f / 384.0f) + 1e-5f);
  float* yr = y + (size_t)row * DIM_C;
#pragma unroll
  for (int i = 0; i < 12; ++i) {
    int c = lane + 32 * i;
    yr[c] = (v[i] - mean) * inv * w[c] + b[c];
  }
}

// ---------------------------------------------------------------------------
// qEh/qEw: [B,H,N,P] = sum_d q[b,i,h*64+d] * rel[p, h*64+d]   (q pre-scaled)
// ---------------------------------------------------------------------------
__global__ void qe_bias_kernel(const float* __restrict__ Q,
                               const float* __restrict__ Eh,
                               const float* __restrict__ Ew,
                               float* __restrict__ qEh, float* __restrict__ qEw,
                               int N, int P) {
  size_t idx = (size_t)blockIdx.x * blockDim.x + threadIdx.x;
  size_t total = (size_t)B_C * HEADS_C * N * P;
  if (idx >= total) return;
  int p = (int)(idx % P); size_t t = idx / P;
  int i = (int)(t % N); t /= N;
  int h = (int)(t % HEADS_C);
  int b = (int)(t / HEADS_C);
  const float* q  = Q  + ((size_t)(b * N + i)) * INNER_C + h * DH_C;
  const float* eh = Eh + (size_t)p * INNER_C + h * DH_C;
  const float* ew = Ew + (size_t)p * INNER_C + h * DH_C;
  float ah = 0.0f, aw = 0.0f;
#pragma unroll 8
  for (int d = 0; d < DH_C; ++d) { ah += q[d] * eh[d]; aw += q[d] * ew[d]; }
  qEh[idx] = ah;
  qEw[idx] = aw;
}

// ---------------------------------------------------------------------------
// Flash attention, one wave per 16 query rows of one (b,h).
// scores = Q@K^T + bias(i,j) gathered from qEh/qEw via clipped 2D buckets.
// ---------------------------------------------------------------------------
__global__ void __launch_bounds__(32)
flash_attn_wmma(const float* __restrict__ Q, const float* __restrict__ Km,
                const float* __restrict__ Vm, float* __restrict__ O,
                const float* __restrict__ qEh, const float* __restrict__ qEw,
                int N, int W, int P) {
  __shared__ _Float16 Pt[16 * 40];
  __shared__ _Float16 Vt[64 * 40];

  const int i0   = blockIdx.x * 16;
  const int h    = blockIdx.y;
  const int b    = blockIdx.z;
  const int lane = threadIdx.x;
  const int hf   = lane >> 4;
  const int l16  = lane & 15;
  const int mclip = (P - 1) >> 1;

  // Q A-fragments (DH=64 -> two K=32 chunks), scale already applied to Q.
  v16h qa[2];
  {
    const float* qbase = Q + ((size_t)(b * N + i0 + l16)) * INNER_C + h * DH_C;
#pragma unroll
    for (int c = 0; c < 2; ++c) {
      const float* p = qbase + c * 32 + 16 * hf;
      v16h r;
#pragma unroll
      for (int e = 0; e < 8; ++e) r[e] = (_Float16)p[e];
#pragma unroll
      for (int e = 0; e < 8; ++e) r[8 + e] = (_Float16)p[8 + e];
      qa[c] = r;
    }
  }

  v8f acc[4];
#pragma unroll
  for (int dt = 0; dt < 4; ++dt) acc[dt] = vzero8();
  float mrow[8], lrow[8];
#pragma unroll
  for (int e = 0; e < 8; ++e) { mrow[e] = -1e30f; lrow[e] = 0.0f; }

  const float* biasH = qEh + ((size_t)(b * HEADS_C + h)) * N * P;
  const float* biasW = qEw + ((size_t)(b * HEADS_C + h)) * N * P;
  int irow[8], icol[8];
#pragma unroll
  for (int e = 0; e < 8; ++e) {
    int i = i0 + e + 8 * hf;
    irow[e] = i / W; icol[e] = i % W;
  }

  for (int j0 = 0; j0 < N; j0 += 32) {
    v8f S[2];
#pragma unroll
    for (int t = 0; t < 2; ++t) {
      const int jt = j0 + 16 * t;
      // K^T B-fragments: lane col n=l16 -> key jt+n, contiguous d along e.
      const float* kbase = Km + ((size_t)(b * N + jt + l16)) * INNER_C + h * DH_C;
      v16h kb[2];
#pragma unroll
      for (int c = 0; c < 2; ++c) {
        const float* p = kbase + c * 32 + 16 * hf;
        v16h r;
#pragma unroll
        for (int e = 0; e < 16; ++e) r[e] = (_Float16)p[e];
        kb[c] = r;
      }
      v8f s = vzero8();
      s = __builtin_amdgcn_wmma_f32_16x16x32_f16(false, qa[0], false, kb[0],
                                                 (short)0, s, false, false);
      s = __builtin_amdgcn_wmma_f32_16x16x32_f16(false, qa[1], false, kb[1],
                                                 (short)0, s, false, false);
      // contextual relative-position bias gather
      int j = jt + l16;
      int rj = j / W, cj = j % W;
#pragma unroll
      for (int e = 0; e < 8; ++e) {
        int dh = irow[e] - rj; dh = dh < -mclip ? -mclip : (dh > mclip ? mclip : dh);
        int dw = icol[e] - cj; dw = dw < -mclip ? -mclip : (dw > mclip ? mclip : dw);
        int i = i0 + e + 8 * hf;
        s[e] += biasH[(size_t)i * P + (dh + mclip)] + biasW[(size_t)i * P + (dw + mclip)];
      }
      S[t] = s;
    }

    // online softmax update
    float alpha_[8];
#pragma unroll
    for (int e = 0; e < 8; ++e) {
      float mx = fmaxf(S[0][e], S[1][e]);
#pragma unroll
      for (int o = 1; o < 16; o <<= 1) mx = fmaxf(mx, __shfl_xor(mx, o, 32));
      float mn = fmaxf(mrow[e], mx);
      alpha_[e] = __expf(mrow[e] - mn);
      mrow[e] = mn;
      float p0 = __expf(S[0][e] - mn);
      float p1 = __expf(S[1][e] - mn);
      S[0][e] = p0; S[1][e] = p1;
      float rs = p0 + p1;
#pragma unroll
      for (int o = 1; o < 16; o <<= 1) rs += __shfl_xor(rs, o, 32);
      lrow[e] = lrow[e] * alpha_[e] + rs;
    }
#pragma unroll
    for (int dt = 0; dt < 4; ++dt)
#pragma unroll
      for (int e = 0; e < 8; ++e) acc[dt][e] *= alpha_[e];

    // P (16x32) to LDS in row-major f16
#pragma unroll
    for (int t = 0; t < 2; ++t)
#pragma unroll
      for (int e = 0; e < 8; ++e)
        Pt[(e + 8 * hf) * 40 + 16 * t + l16] = (_Float16)S[t][e];

    // stage V tile transposed: Vt[d][k], k = key - j0
    {
      const float* vrow = Vm + ((size_t)(b * N + j0 + lane)) * INNER_C + h * DH_C;
#pragma unroll 8
      for (int d = 0; d < DH_C; ++d) Vt[d * 40 + lane] = (_Float16)vrow[d];
    }
    __syncthreads();

    v16h pa = ldsA_f16(Pt, 40);
#pragma unroll
    for (int dt = 0; dt < 4; ++dt) {
      v16h vb = ldsB_f16(&Vt[(dt * 16) * 40], 40);
      acc[dt] = __builtin_amdgcn_wmma_f32_16x16x32_f16(false, pa, false, vb,
                                                       (short)0, acc[dt], false, false);
    }
    __syncthreads();
  }

  // finalize: O = acc / l
  float* obase = O + ((size_t)(b * N + i0)) * INNER_C + h * DH_C;
#pragma unroll
  for (int dt = 0; dt < 4; ++dt) {
#pragma unroll
    for (int e = 0; e < 8; ++e) {
      int r = e + 8 * hf;
      obase[(size_t)r * INNER_C + dt * 16 + l16] = acc[dt][e] / lrow[e];
    }
  }
}

// ---------------------------------------------------------------------------
// im2col for 3x3 stride-2 pad-1 conv from token layout (B, Hh*Ww, DIM).
// col[b][oy*OW+ox][ci*9 + ky*3 + kx]
// ---------------------------------------------------------------------------
__global__ void im2col_kernel(const float* __restrict__ t, float* __restrict__ col,
                              int Hh, int Ww) {
  int OH = Hh >> 1, OW = Ww >> 1;
  size_t total = (size_t)B_C * OH * OW * DIM_C * 9;
  size_t idx = (size_t)blockIdx.x * blockDim.x + threadIdx.x;
  if (idx >= total) return;
  int kk = (int)(idx % 9); size_t t1 = idx / 9;
  int ci = (int)(t1 % DIM_C); t1 /= DIM_C;
  int ox = (int)(t1 % OW); t1 /= OW;
  int oy = (int)(t1 % OH);
  int b  = (int)(t1 / OH);
  int ky = kk / 3, kx = kk % 3;
  int py = 2 * oy - 1 + ky, px = 2 * ox - 1 + kx;
  float v = 0.0f;
  if (py >= 0 && py < Hh && px >= 0 && px < Ww)
    v = t[((size_t)b * Hh * Ww + (size_t)py * Ww + px) * DIM_C + ci];
  col[idx] = v;
}

// x (B, CIN, 32, 32) -> xt (B, 1024, CIN)
__global__ void perm_in_kernel(const float* __restrict__ x, float* __restrict__ xt) {
  size_t total = (size_t)B_C * CIN_C * 1024;
  size_t idx = (size_t)blockIdx.x * blockDim.x + threadIdx.x;
  if (idx >= total) return;
  int c = (int)(idx % CIN_C); size_t t = idx / CIN_C;
  int p = (int)(t % 1024);
  int b = (int)(t / 1024);
  xt[idx] = x[((size_t)b * CIN_C + c) * 1024 + p];
}

// t (B, 64, COUT) -> out (B, COUT, 8, 8)
__global__ void perm_out_kernel(const float* __restrict__ t, float* __restrict__ out) {
  size_t total = (size_t)B_C * COUT_C * 64;
  size_t idx = (size_t)blockIdx.x * blockDim.x + threadIdx.x;
  if (idx >= total) return;
  int p = (int)(idx % 64); size_t t1 = idx / 64;
  int c = (int)(t1 % COUT_C);
  int b = (int)(t1 / COUT_C);
  out[idx] = t[((size_t)b * 64 + p) * COUT_C + c];
}

// ---------------------------------------------------------------------------
// Host-side orchestration
// ---------------------------------------------------------------------------
struct BlkP {
  const float *ln1_w, *ln1_b, *wq, *wk, *wv, *wo, *ln2_w, *ln2_b, *w1, *b1, *w2, *b2;
};

static inline void launch_gemm(bool bt, const float* A, const float* B, float* C,
                               const float* bias, const float* res,
                               int M, int N, int K, float alpha, int act,
                               hipStream_t s) {
  dim3 grid(N / 64, M / 64), blk(128);
  if (bt)
    gemm_wmma_kernel<true><<<grid, blk, 0, s>>>(A, B, C, bias, res, M, N, K, alpha, act);
  else
    gemm_wmma_kernel<false><<<grid, blk, 0, s>>>(A, B, C, bias, res, M, N, K, alpha, act);
}

extern "C" void kernel_launch(void* const* d_in, const int* in_sizes, int n_in,
                              void* d_out, int out_size, void* d_ws, size_t ws_size,
                              hipStream_t stream) {
  const int BUCK[3] = {33, 17, 9};

  // ---- resolve input pointers (handle both plausible flattening orders) ----
  const float* x = nullptr;
  const float *embed_w, *embed_b, *head_ln_w, *head_ln_b, *head_w, *head_b;
  const float *conv_w[2], *conv_b[2];
  const float *rel_h[3], *rel_w[3];
  BlkP blk[3][2];
  const float** F = (const float**)d_in;

  bool xfirst = (in_sizes[0] == B_C * CIN_C * 32 * 32);
  if (xfirst) {
    // insertion-order recursive flatten: x, then params in insertion order
    x = F[0];
    embed_w = F[1]; embed_b = F[2];
    head_ln_w = F[3]; head_ln_b = F[4]; head_w = F[5]; head_b = F[6];
    conv_w[0] = F[7]; conv_b[0] = F[8]; conv_w[1] = F[9]; conv_b[1] = F[10];
    for (int s = 0; s < 3; ++s) {
      int base = 11 + 26 * s;
      for (int k = 0; k < 2; ++k) {
        int o = base + 12 * k;
        blk[s][k].ln1_w = F[o + 0];  blk[s][k].ln1_b = F[o + 1];
        blk[s][k].wq    = F[o + 2];  blk[s][k].wk    = F[o + 3];
        blk[s][k].wv    = F[o + 4];  blk[s][k].wo    = F[o + 5];
        blk[s][k].ln2_w = F[o + 6];  blk[s][k].ln2_b = F[o + 7];
        blk[s][k].w1    = F[o + 8];  blk[s][k].b1    = F[o + 9];
        blk[s][k].w2    = F[o + 10]; blk[s][k].b2    = F[o + 11];
      }
      rel_h[s] = F[base + 24]; rel_w[s] = F[base + 25];
    }
  } else {
    // jax pytree (sorted dict keys): params first, x last
    conv_b[0] = F[0]; conv_w[0] = F[1]; conv_b[1] = F[2]; conv_w[1] = F[3];
    embed_b = F[4]; embed_w = F[5];
    head_b = F[6]; head_ln_b = F[7]; head_ln_w = F[8]; head_w = F[9];
    for (int s = 0; s < 3; ++s) {
      int base = 10 + 26 * s;
      for (int k = 0; k < 2; ++k) {
        int o = base + 12 * k;
        blk[s][k].b1    = F[o + 0];  blk[s][k].b2    = F[o + 1];
        blk[s][k].ln1_b = F[o + 2];  blk[s][k].ln1_w = F[o + 3];
        blk[s][k].ln2_b = F[o + 4];  blk[s][k].ln2_w = F[o + 5];
        blk[s][k].w1    = F[o + 6];  blk[s][k].w2    = F[o + 7];
        blk[s][k].wk    = F[o + 8];  blk[s][k].wo    = F[o + 9];
        blk[s][k].wq    = F[o + 10]; blk[s][k].wv    = F[o + 11];
      }
      rel_h[s] = F[base + 24]; rel_w[s] = F[base + 25];
    }
    x = F[10 + 26 * 3];
  }

  // ---- workspace carve-up (floats) ----
  float* ws = (float*)d_ws;
  const size_t TOK = (size_t)B_C * 1024 * DIM_C;   // 1,572,864
  float* XT   = ws;             ws += TOK;
  float* X    = ws;             ws += TOK;
  float* H    = ws;             ws += TOK;
  float* Qb   = ws;             ws += TOK;
  float* Kb   = ws;             ws += TOK;
  float* Vb   = ws;             ws += TOK;
  float* Ob   = ws;             ws += TOK;
  float* MLPH = ws;             ws += (size_t)B_C * 1024 * MLP_C;
  float* QEH  = ws;             ws += (size_t)B_C * HEADS_C * 1024 * 33;
  float* QEW  = ws;             ws += (size_t)B_C * HEADS_C * 1024 * 33;
  float* COL  = ws;             ws += (size_t)B_C * 256 * DIM_C * 9;
  float* HOUT = ws;             ws += (size_t)B_C * 64 * COUT_C;

  const float qscale = 0.125f;  // 1/sqrt(64)

  // ---- embed ----
  {
    size_t tot = (size_t)B_C * CIN_C * 1024;
    perm_in_kernel<<<dim3((tot + 255) / 256), dim3(256), 0, stream>>>(x, XT);
    launch_gemm(false, XT, embed_w, X, embed_b, nullptr,
                B_C * 1024, DIM_C, CIN_C, 1.0f, 0, stream);
  }

  // ---- stages ----
  int hh = 32, ww = 32;
  for (int s = 0; s < 3; ++s) {
    const int n = hh * ww;
    const int M = B_C * n;
    const int P = BUCK[s];
    for (int k = 0; k < 2; ++k) {
      const BlkP& p = blk[s][k];
      ln_fwd_kernel<<<dim3((M + 3) / 4), dim3(128), 0, stream>>>(X, H, p.ln1_w, p.ln1_b, M);
      launch_gemm(false, H, p.wq, Qb, nullptr, nullptr, M, INNER_C, DIM_C, qscale, 0, stream);
      launch_gemm(false, H, p.wk, Kb, nullptr, nullptr, M, INNER_C, DIM_C, 1.0f, 0, stream);
      launch_gemm(false, H, p.wv, Vb, nullptr, nullptr, M, INNER_C, DIM_C, 1.0f, 0, stream);
      if (k == 0) {
        size_t tot = (size_t)B_C * HEADS_C * n * P;
        qe_bias_kernel<<<dim3((tot + 255) / 256), dim3(256), 0, stream>>>(
            Qb, rel_h[s], rel_w[s], QEH, QEW, n, P);
      }
      flash_attn_wmma<<<dim3(n / 16, HEADS_C, B_C), dim3(32), 0, stream>>>(
          Qb, Kb, Vb, Ob, QEH, QEW, n, ww, P);
      launch_gemm(false, Ob, p.wo, X, nullptr, X, M, DIM_C, INNER_C, 1.0f, 0, stream);
      ln_fwd_kernel<<<dim3((M + 3) / 4), dim3(128), 0, stream>>>(X, H, p.ln2_w, p.ln2_b, M);
      launch_gemm(false, H, p.w1, MLPH, p.b1, nullptr, M, MLP_C, DIM_C, 1.0f, 1, stream);
      launch_gemm(false, MLPH, p.w2, X, p.b2, X, M, DIM_C, MLP_C, 1.0f, 0, stream);
    }
    if (s < 2) {
      size_t tot = (size_t)B_C * (hh / 2) * (ww / 2) * DIM_C * 9;
      im2col_kernel<<<dim3((tot + 255) / 256), dim3(256), 0, stream>>>(X, COL, hh, ww);
      hh >>= 1; ww >>= 1;
      launch_gemm(true, COL, conv_w[s], X, conv_b[s], nullptr,
                  B_C * hh * ww, DIM_C, DIM_C * 9, 1.0f, 0, stream);
    }
  }

  // ---- head ----
  {
    const int M = B_C * 64;
    ln_fwd_kernel<<<dim3((M + 3) / 4), dim3(128), 0, stream>>>(X, H, head_ln_w, head_ln_b, M);
    launch_gemm(false, H, head_w, HOUT, head_b, nullptr, M, COUT_C, DIM_C, 1.0f, 0, stream);
    size_t tot = (size_t)B_C * COUT_C * 64;
    perm_out_kernel<<<dim3((tot + 255) / 256), dim3(256), 0, stream>>>(HOUT, (float*)d_out);
  }

  (void)n_in; (void)out_size; (void)ws_size; (void)in_sizes;
}
